// VRWKV_SpatialMix_V6_85066122264797
// MI455X (gfx1250) — matrix operand order repaired and checked by reading, not compiled, not run on hardware
//
#include <hip/hip_runtime.h>
#include <hip/hip_bf16.h>

// ---------------- problem constants (fixed by setup_inputs) ----------------
#define B_   8
#define T_   1024
#define C_   768
#define H_   24
#define NH_  32          // head size
#define G_   2
#define CG_  384         // C per group
#define BT_  (B_ * T_)   // 8192 tokens
#define PH_  32
#define PW_  32

typedef __attribute__((ext_vector_type(16))) _Float16 v16h;
typedef __attribute__((ext_vector_type(8)))  float    v8f;

enum { EPI_STORE = 0, EPI_TANH = 1, EPI_BIAS = 2 };

// ===========================================================================
// WMMA GEMM:  Y[n][m] = epi( sum_k A[m][k] * B[n][k] )
//   A : f16 weights, row-major (M x K), lda = K stride
//   B : f16 activations, token-major [n][k], row stride ldb, column offset bcol
//   Y : OutT token-major [n][m], row stride ldy, column offset ycol
// Wave tile: 32(M) x 64(N).  Per 32-K step: 2 A frags + 4 B frags -> 8 WMMAs
// issued back-to-back after all loads are in flight (single basic block, so
// the accumulators stay pinned in-place across iterations).
// Fragment K order is contiguous: lane half h holds K 16h..16h+15 for A and B (wave32).
// ===========================================================================
template<int EPI, typename OutT>
__global__ __launch_bounds__(128)
void gemm_wmma_f16(const _Float16* __restrict__ A, int lda,
                   const _Float16* __restrict__ Bm, int ldb, int bcol,
                   OutT* __restrict__ Y, int ldy, int ycol,
                   const float* __restrict__ bias,
                   int M, int N, int K)
{
    const int lane = threadIdx.x & 31;
    const int wave = threadIdx.x >> 5;
    const int m0 = blockIdx.y * 32;
    const int n0 = (blockIdx.x * 4 + wave) * 64;
    if (m0 >= M || n0 >= N) return;

    v8f acc00 = {}, acc01 = {}, acc02 = {}, acc03 = {};
    v8f acc10 = {}, acc11 = {}, acc12 = {}, acc13 = {};

    // A fragment addressing: lanes 0-15 -> M rows 0-15 (halfsel=0 covers K
    // 0..15 of the 32-K step; halfsel=1 covers 16..31).
    const int ahalf = (lane >> 4) * 16;
    const _Float16* pa0 = A + (size_t)(m0 +      (lane & 15)) * lda + ahalf;
    const _Float16* pa1 = A + (size_t)(m0 + 16 + (lane & 15)) * lda + ahalf;
    // B fragment addressing: n = n0 + lane%16 ; lanes 0-15 K 0-15, 16-31 K 16-31.
    const int bko = (lane >> 4) * 16;
    const _Float16* pb0 = Bm + (size_t)(n0 +      (lane & 15)) * ldb + bcol + bko;
    const _Float16* pb1 = Bm + (size_t)(n0 + 16 + (lane & 15)) * ldb + bcol + bko;
    const _Float16* pb2 = Bm + (size_t)(n0 + 32 + (lane & 15)) * ldb + bcol + bko;
    const _Float16* pb3 = Bm + (size_t)(n0 + 48 + (lane & 15)) * ldb + bcol + bko;

    for (int k0 = 0; k0 < K; k0 += 32) {
        // ---- issue all fragment loads for this K step first ----
        v16h a0, a1, b0, b1, b2, b3;
#pragma unroll
        for (int i = 0; i < 16; ++i) a0[i] = pa0[k0 + i];
#pragma unroll
        for (int i = 0; i < 16; ++i) a1[i] = pa1[k0 + i];
#pragma unroll
        for (int i = 0; i < 16; ++i) b0[i] = pb0[k0 + i];
#pragma unroll
        for (int i = 0; i < 16; ++i) b1[i] = pb1[k0 + i];
#pragma unroll
        for (int i = 0; i < 16; ++i) b2[i] = pb2[k0 + i];
#pragma unroll
        for (int i = 0; i < 16; ++i) b3[i] = pb3[k0 + i];
        // Speculative prefetch of the next K step (silently dropped past end).
        __builtin_prefetch(pb0 + k0 + 32, 0, 1);
        __builtin_prefetch(pb1 + k0 + 32, 0, 1);
        __builtin_prefetch(pb2 + k0 + 32, 0, 1);
        __builtin_prefetch(pb3 + k0 + 32, 0, 1);

        // ---- 8 back-to-back WMMAs (A frags reused across 4 B frags) ----
        acc00 = __builtin_amdgcn_wmma_f32_16x16x32_f16(false, a0, false, b0, (short)0, acc00, false, false);
        acc01 = __builtin_amdgcn_wmma_f32_16x16x32_f16(false, a0, false, b1, (short)0, acc01, false, false);
        acc02 = __builtin_amdgcn_wmma_f32_16x16x32_f16(false, a0, false, b2, (short)0, acc02, false, false);
        acc03 = __builtin_amdgcn_wmma_f32_16x16x32_f16(false, a0, false, b3, (short)0, acc03, false, false);
        acc10 = __builtin_amdgcn_wmma_f32_16x16x32_f16(false, a1, false, b0, (short)0, acc10, false, false);
        acc11 = __builtin_amdgcn_wmma_f32_16x16x32_f16(false, a1, false, b1, (short)0, acc11, false, false);
        acc12 = __builtin_amdgcn_wmma_f32_16x16x32_f16(false, a1, false, b2, (short)0, acc12, false, false);
        acc13 = __builtin_amdgcn_wmma_f32_16x16x32_f16(false, a1, false, b3, (short)0, acc13, false, false);
    }

    // D layout: lane -> token n + lane%16, VGPR r -> channel m0 + (lane/16)*8 + r.
    const int mo = (lane >> 4) * 8;
    const int nr = lane & 15;
    v8f* accs[8] = { &acc00, &acc01, &acc02, &acc03, &acc10, &acc11, &acc12, &acc13 };
#pragma unroll
    for (int mt = 0; mt < 2; ++mt) {
#pragma unroll
        for (int nt = 0; nt < 4; ++nt) {
            v8f a = *accs[mt * 4 + nt];
            OutT* py = Y + (size_t)(n0 + nt * 16 + nr) * ldy
                         + ycol + m0 + mt * 16 + mo;
#pragma unroll
            for (int r = 0; r < 8; ++r) {
                float v = a[r];
                if (EPI == EPI_TANH) v = tanhf(v);
                if (EPI == EPI_BIAS) v += bias[m0 + mt * 16 + mo + r];
                py[r] = (OutT)v;
            }
        }
    }
}

template<int EPI, typename OutT>
static void launch_gemm(const _Float16* A, int lda,
                        const _Float16* Bm, int ldb, int bcol,
                        OutT* Y, int ldy, int ycol, const float* bias,
                        int M, int N, int K, hipStream_t s)
{
    dim3 grid(N / 256, M / 32);     // all M here are multiples of 32
    hipLaunchKernelGGL((gemm_wmma_f16<EPI, OutT>), grid, dim3(128), 0, s,
                       A, lda, Bm, ldb, bcol, Y, ldy, ycol, bias, M, N, K);
}

// ---------------- weight conversion helpers (f32 -> f16, opt. transpose) ---
__global__ void cvt_f16_kernel(const float* __restrict__ s,
                               _Float16* __restrict__ d, int n)
{
    int i = blockIdx.x * blockDim.x + threadIdx.x;
    if (i < n) d[i] = (_Float16)s[i];
}

// s: batch x (rows x cols) -> d: batch x (cols x rows)
__global__ void cvtT_f16_kernel(const float* __restrict__ s,
                                _Float16* __restrict__ d,
                                int rows, int cols, int batch)
{
    int i = blockIdx.x * blockDim.x + threadIdx.x;
    int per = rows * cols;
    if (i >= per * batch) return;
    int bi = i / per, rem = i % per;
    int r = rem / cols, c = rem % cols;
    d[(size_t)bi * per + (size_t)c * rows + r] = (_Float16)s[i];
}

// ---------------- q_shift + token-mix GEMM inputs --------------------------
__global__ void prep_kernel(const float* __restrict__ x,
                            const float* __restrict__ xcnd,
                            const float* __restrict__ tmx,
                            const float* __restrict__ tmxc,
                            float* __restrict__ xx, float* __restrict__ xxc,
                            _Float16* __restrict__ Ain1,
                            _Float16* __restrict__ Ain2)
{
    size_t i = (size_t)blockIdx.x * blockDim.x + threadIdx.x;
    if (i >= (size_t)BT_ * C_) return;
    int c = (int)(i % C_);
    int n = (int)(i / C_);
    int t = n & (T_ - 1);
    int b = n >> 10;
    int hh = t >> 5, ww = t & 31;
    int q = (c & (NH_ - 1)) >> 3;            // quarter within head
    int sh = hh, sw = ww; bool ok;
    if      (q == 0) { sw = ww - 1; ok = (sw >= 0);  }   // shift right
    else if (q == 1) { sw = ww + 1; ok = (sw < PW_); }   // shift left
    else if (q == 2) { sh = hh - 1; ok = (sh >= 0);  }   // shift down
    else             { sh = hh + 1; ok = (sh < PH_); }   // shift up
    size_t sidx = ((size_t)b * T_ + sh * PW_ + sw) * C_ + c;

    float xv = x[i];
    float d0 = (ok ? x[sidx] : 0.f) - xv;
    xx[i] = d0;
    Ain1[i] = (_Float16)(xv + d0 * tmx[c]);

    float xcv = xcnd[i];
    float d1 = (ok ? xcnd[sidx] : 0.f) - xcv;
    xxc[i] = d1;
    Ain2[i] = (_Float16)(xcv + d1 * tmxc[c]);
}

// ---------------- faithful replica of the (4,B,32,T)->(4,B*T,32) reshape ---
// ap[j][b*T + t2][s] = a[b][j*32 + (t2>>5)][(t2&31)*32 + s]
// where a_tok is token-major (B*T, 128).
__global__ void repack_a_kernel(const _Float16* __restrict__ a_tok,
                                _Float16* __restrict__ ap)
{
    int i = blockIdx.x * blockDim.x + threadIdx.x;     // < 4*BT*32
    if (i >= 4 * BT_ * 32) return;
    int s  = i & 31;
    int nj = i >> 5;
    int n  = nj & (BT_ - 1);
    int j  = nj >> 13;
    int b  = n >> 10;
    int t2 = n & (T_ - 1);
    int src_tok = b * T_ + ((t2 & 31) << 5) + s;
    int src_ch  = j * 32 + (t2 >> 5);
    ap[i] = a_tok[(size_t)src_tok * 128 + src_ch];
}

// ---------------- build xw/xk/xv/xr/xg (f16, token-major) ------------------
__global__ void combine_kernel(const float* __restrict__ x,
                               const float* __restrict__ xcnd,
                               const float* __restrict__ xx,
                               const float* __restrict__ xxc,
                               const _Float16* __restrict__ mw,
                               const _Float16* __restrict__ mk,
                               const _Float16* __restrict__ mr,
                               const _Float16* __restrict__ mg,
                               const _Float16* __restrict__ mvp,
                               const float* __restrict__ tw,
                               const float* __restrict__ tk,
                               const float* __restrict__ tv,
                               const float* __restrict__ tr,
                               const float* __restrict__ tg,
                               _Float16* __restrict__ Xw, _Float16* __restrict__ Xk,
                               _Float16* __restrict__ Xv, _Float16* __restrict__ Xr,
                               _Float16* __restrict__ Xg)
{
    size_t i = (size_t)blockIdx.x * blockDim.x + threadIdx.x;
    if (i >= (size_t)BT_ * C_) return;
    int c = (int)(i % C_);
    float xv_ = x[i],   dxv = xx[i];
    float xcv = xcnd[i], dxc = xxc[i];
    float fmk = (float)mk[i];
    float fmv = ((float)mvp[i] + fmk) * 0.5f;
    Xw[i] = (_Float16)(xv_ + dxv * (tw[c] + (float)mw[i]));
    Xk[i] = (_Float16)(xv_ + dxv * (tk[c] + fmk));
    Xv[i] = (_Float16)(xcv + dxc * (tv[c] + fmv));
    Xr[i] = (_Float16)(xv_ + dxv * (tr[c] + (float)mr[i]));
    Xg[i] = (_Float16)(xv_ + dxv * (tg[c] + (float)mg[i]));
}

// ---------------- WKV6 recurrence: one wave per (b,h) ----------------------
// Lane i holds state column S[j][i], j=0..31 in registers. Per step:
//   y_i = sum_j r_j * (S_ji + u_j * k_j * v_i) ;  S_ji = S_ji * d_j + k_j * v_i
__global__ __launch_bounds__(32)
void wkv6_kernel(const float* __restrict__ r, const float* __restrict__ k,
                 const float* __restrict__ v, const float* __restrict__ w,
                 const float* __restrict__ u, float* __restrict__ y)
{
    int bh = blockIdx.x;                // 0..B*H-1
    int b = bh / H_, h = bh % H_;
    int lane = threadIdx.x;
    float S[NH_];
#pragma unroll
    for (int j = 0; j < NH_; ++j) S[j] = 0.f;
    float ui = u[h * NH_ + lane];
    size_t base = (size_t)b * T_ * C_ + h * NH_ + lane;
    for (int t = 0; t < T_; ++t) {
        size_t idx = base + (size_t)t * C_;
        float ri = r[idx], ki = k[idx], vi = v[idx];
        float di = __expf(-__expf(w[idx]));
        float yi = 0.f;
#pragma unroll
        for (int j = 0; j < NH_; ++j) {
            float rj = __shfl(ri, j, 32);
            float kj = __shfl(ki, j, 32);
            float dj = __shfl(di, j, 32);
            float uj = __shfl(ui, j, 32);
            float kv = kj * vi;
            yi += rj * (S[j] + uj * kv);
            S[j] = S[j] * dj + kv;
        }
        y[idx] = yi;
    }
}

// ---------------- LayerNorm over C + SiLU gate -> f16 Z --------------------
__global__ __launch_bounds__(256)
void ln_gate_kernel(const float* __restrict__ y, const float* __restrict__ g,
                    const float* __restrict__ gamma, const float* __restrict__ beta,
                    _Float16* __restrict__ z)
{
    __shared__ float ss[8], ss2[8];
    int n = blockIdx.x;
    const float* yr = y + (size_t)n * C_;
    float s = 0.f, s2 = 0.f;
    for (int c = threadIdx.x; c < C_; c += 256) {
        float t = yr[c]; s += t; s2 += t * t;
    }
    for (int o = 16; o > 0; o >>= 1) {
        s  += __shfl_down(s,  o, 32);
        s2 += __shfl_down(s2, o, 32);
    }
    int wv = threadIdx.x >> 5, ln = threadIdx.x & 31;
    if (!ln) { ss[wv] = s; ss2[wv] = s2; }
    __syncthreads();
    if (threadIdx.x == 0) {
        float a = 0.f, b2 = 0.f;
        for (int i = 0; i < 8; ++i) { a += ss[i]; b2 += ss2[i]; }
        float m = a / C_;
        ss[0] = m; ss2[0] = b2 / C_ - m * m;
    }
    __syncthreads();
    float mean = ss[0];
    float inv  = rsqrtf(ss2[0] + 1e-5f);
    const float* gr = g + (size_t)n * C_;
    for (int c = threadIdx.x; c < C_; c += 256) {
        float yl = (yr[c] - mean) * inv * gamma[c] + beta[c];
        float gv = gr[c];
        float sil = gv / (1.f + __expf(-gv));
        z[(size_t)n * C_ + c] = (_Float16)(yl * sil);
    }
}

// ===========================================================================
extern "C" void kernel_launch(void* const* d_in, const int* in_sizes, int n_in,
                              void* d_out, int out_size, void* d_ws, size_t ws_size,
                              hipStream_t stream)
{
    (void)in_sizes; (void)n_in; (void)out_size; (void)ws_size;

    const float* x      = (const float*)d_in[0];
    const float* xcnd   = (const float*)d_in[1];
    const float* tmx    = (const float*)d_in[4];
    const float* tmxc   = (const float*)d_in[5];
    const float* tmw    = (const float*)d_in[6];
    const float* tmk    = (const float*)d_in[7];
    const float* tmv    = (const float*)d_in[8];
    const float* tmr    = (const float*)d_in[9];
    const float* tmg    = (const float*)d_in[10];
    const float* w1     = (const float*)d_in[11];  // (128,768)
    const float* w2     = (const float*)d_in[12];  // (4,32,768)
    const float* w1c    = (const float*)d_in[13];  // (32,768)
    const float* w2c    = (const float*)d_in[14];  // (32,768)
    const float* tdecay = (const float*)d_in[15];  // (C)
    const float* dw1    = (const float*)d_in[16];  // (64,768)
    const float* dw2    = (const float*)d_in[17];  // (768,64)
    const float* faaaa  = (const float*)d_in[18];  // (H,N)
    const float* key_w  = (const float*)d_in[19];  // (2,384,384)
    const float* val_w  = (const float*)d_in[20];
    const float* rec_w  = (const float*)d_in[21];
    const float* gat_w  = (const float*)d_in[22];
    const float* out_w  = (const float*)d_in[23];
    const float* gamma  = (const float*)d_in[24];
    const float* beta   = (const float*)d_in[25];
    float* out = (float*)d_out;

    // ---- workspace arena ----
    const size_t F32 = (size_t)BT_ * C_ * sizeof(float);
    const size_t F16 = (size_t)BT_ * C_ * sizeof(_Float16);
    char* p = (char*)d_ws;
    auto take = [&](size_t bytes) {
        void* r = (void*)p;
        p += (bytes + 255) & ~(size_t)255;
        return r;
    };
    float* xxb  = (float*)take(F32);
    float* xxcb = (float*)take(F32);
    float* rbuf = (float*)take(F32);
    float* kbuf = (float*)take(F32);
    float* vbuf = (float*)take(F32);
    float* gbuf = (float*)take(F32);
    float* wbuf = (float*)take(F32);
    _Float16* Ain1 = (_Float16*)take(F16);
    _Float16* Ain2 = (_Float16*)take(F16);
    _Float16* mwb  = (_Float16*)take(F16);
    _Float16* mkb  = (_Float16*)take(F16);
    _Float16* mrb  = (_Float16*)take(F16);
    _Float16* mgb  = (_Float16*)take(F16);
    _Float16* mvp  = (_Float16*)take(F16);
    _Float16* Xw   = (_Float16*)take(F16);
    _Float16* Xk   = (_Float16*)take(F16);
    _Float16* Xv   = (_Float16*)take(F16);
    _Float16* Xr   = (_Float16*)take(F16);
    _Float16* Xg   = (_Float16*)take(F16);
    _Float16* a_tok  = (_Float16*)take((size_t)BT_ * 128 * 2);
    _Float16* ap     = (_Float16*)take((size_t)4 * BT_ * 32 * 2);
    _Float16* bcross = (_Float16*)take((size_t)BT_ * 32 * 2);
    _Float16* hdec   = (_Float16*)take((size_t)BT_ * 64 * 2);
    _Float16* w1h  = (_Float16*)take(128 * 768 * 2);
    _Float16* w2t  = (_Float16*)take(4 * 768 * 32 * 2);
    _Float16* w1ch = (_Float16*)take(32 * 768 * 2);
    _Float16* w2ct = (_Float16*)take(768 * 32 * 2);
    _Float16* dw1h = (_Float16*)take(64 * 768 * 2);
    _Float16* dw2h = (_Float16*)take(768 * 64 * 2);
    _Float16* kwh  = (_Float16*)take((size_t)G_ * CG_ * CG_ * 2);
    _Float16* vwh  = (_Float16*)take((size_t)G_ * CG_ * CG_ * 2);
    _Float16* rwh  = (_Float16*)take((size_t)G_ * CG_ * CG_ * 2);
    _Float16* gwh  = (_Float16*)take((size_t)G_ * CG_ * CG_ * 2);
    _Float16* owh  = (_Float16*)take((size_t)G_ * CG_ * CG_ * 2);
    // safe aliases (lifetimes disjoint): y reuses xx, Z reuses Ain1
    float*    ybuf = xxb;
    _Float16* Z    = Ain1;

    auto cvt = [&](const float* s, _Float16* d, int n) {
        hipLaunchKernelGGL(cvt_f16_kernel, dim3((n + 255) / 256), dim3(256), 0,
                           stream, s, d, n);
    };

    // ---- 0) weight conversions (f32 -> f16, transposed where A needs M x K)
    cvt(w1,  w1h,  128 * 768);
    hipLaunchKernelGGL(cvtT_f16_kernel, dim3((4 * 32 * 768 + 255) / 256),
                       dim3(256), 0, stream, w2, w2t, 32, 768, 4);
    cvt(w1c, w1ch, 32 * 768);
    hipLaunchKernelGGL(cvtT_f16_kernel, dim3((32 * 768 + 255) / 256),
                       dim3(256), 0, stream, w2c, w2ct, 32, 768, 1);
    cvt(dw1, dw1h, 64 * 768);
    cvt(dw2, dw2h, 768 * 64);
    cvt(key_w, kwh, G_ * CG_ * CG_);
    cvt(val_w, vwh, G_ * CG_ * CG_);
    cvt(rec_w, rwh, G_ * CG_ * CG_);
    cvt(gat_w, gwh, G_ * CG_ * CG_);
    cvt(out_w, owh, G_ * CG_ * CG_);

    // ---- 1) q_shift + mix-GEMM inputs
    {
        size_t tot = (size_t)BT_ * C_;
        hipLaunchKernelGGL(prep_kernel, dim3((unsigned)((tot + 255) / 256)),
                           dim3(256), 0, stream,
                           x, xcnd, tmx, tmxc, xxb, xxcb, Ain1, Ain2);
    }

    // ---- 2) a = tanh(W1 * mix)  : M=128 K=768 N=8192
    launch_gemm<EPI_TANH, _Float16>(w1h, C_, Ain1, C_, 0, a_tok, 128, 0,
                                    nullptr, 128, BT_, C_, stream);
    // ---- 3) faithful reshape permutation
    hipLaunchKernelGGL(repack_a_kernel, dim3((4 * BT_ * 32 + 255) / 256),
                       dim3(256), 0, stream, a_tok, ap);
    // ---- 4) m_j = w2t[j] * ap[j] : M=768 K=32
    _Float16* mouts[4] = { mwb, mkb, mrb, mgb };
    for (int j = 0; j < 4; ++j)
        launch_gemm<EPI_STORE, _Float16>(w2t + (size_t)j * 768 * 32, 32,
                                         ap + (size_t)j * BT_ * 32, 32, 0,
                                         mouts[j], C_, 0, nullptr,
                                         C_, BT_, 32, stream);
    // ---- 5) cross path: b = tanh(w1_cross * mix_cond), mvp = w2_cross^T * b
    launch_gemm<EPI_TANH, _Float16>(w1ch, C_, Ain2, C_, 0, bcross, 32, 0,
                                    nullptr, 32, BT_, C_, stream);
    launch_gemm<EPI_STORE, _Float16>(w2ct, 32, bcross, 32, 0, mvp, C_, 0,
                                     nullptr, C_, BT_, 32, stream);
    // ---- 6) combine into xw/xk/xv/xr/xg
    {
        size_t tot = (size_t)BT_ * C_;
        hipLaunchKernelGGL(combine_kernel, dim3((unsigned)((tot + 255) / 256)),
                           dim3(256), 0, stream,
                           x, xcnd, xxb, xxcb, mwb, mkb, mrb, mgb, mvp,
                           tmw, tmk, tmv, tmr, tmg, Xw, Xk, Xv, Xr, Xg);
    }
    // ---- 7) grouped convs (r, k, v, gate) : per group M=384 K=384 N=8192
    for (int g = 0; g < G_; ++g) {
        size_t wo = (size_t)g * CG_ * CG_;
        int co = g * CG_;
        launch_gemm<EPI_STORE, float>(rwh + wo, CG_, Xr, C_, co, rbuf, C_, co,
                                      nullptr, CG_, BT_, CG_, stream);
        launch_gemm<EPI_STORE, float>(kwh + wo, CG_, Xk, C_, co, kbuf, C_, co,
                                      nullptr, CG_, BT_, CG_, stream);
        launch_gemm<EPI_STORE, float>(vwh + wo, CG_, Xv, C_, co, vbuf, C_, co,
                                      nullptr, CG_, BT_, CG_, stream);
        launch_gemm<EPI_STORE, float>(gwh + wo, CG_, Xg, C_, co, gbuf, C_, co,
                                      nullptr, CG_, BT_, CG_, stream);
    }
    // ---- 8) decay MLP: h = tanh(dw1 * xw); w = time_decay + dw2 * h
    launch_gemm<EPI_TANH, _Float16>(dw1h, C_, Xw, C_, 0, hdec, 64, 0,
                                    nullptr, 64, BT_, C_, stream);
    launch_gemm<EPI_BIAS, float>(dw2h, 64, hdec, 64, 0, wbuf, C_, 0,
                                 tdecay, C_, BT_, 64, stream);
    // ---- 9) WKV6 recurrence
    hipLaunchKernelGGL(wkv6_kernel, dim3(B_ * H_), dim3(32), 0, stream,
                       rbuf, kbuf, vbuf, wbuf, faaaa, ybuf);
    // ---- 10) LayerNorm + SiLU gate -> Z (f16)
    hipLaunchKernelGGL(ln_gate_kernel, dim3(BT_), dim3(256), 0, stream,
                       ybuf, gbuf, gamma, beta, Z);
    // ---- 11) output grouped conv -> d_out (token-major == (B,T,C))
    for (int g = 0; g < G_; ++g) {
        size_t wo = (size_t)g * CG_ * CG_;
        int co = g * CG_;
        launch_gemm<EPI_STORE, float>(owh + wo, CG_, Z, C_, co, out, C_, co,
                                      nullptr, CG_, BT_, CG_, stream);
    }
}
